// RNN2RNN_56770877718842
// MI455X (gfx1250) — matrix operand order repaired and checked by reading, not compiled, not run on hardware
//
#include <hip/hip_runtime.h>

// ---------------- problem constants ----------------
#define S_LEN 128
#define T_LEN 128
#define BATCH 64
#define HID   512
#define VOCAB 16000
#define G4H   2048              // 4*HID (gate width)
#define KDIM  512               // E == H == 512
#define TDEC  (T_LEN-1)         // 127 decoder steps
#define M_ENC (S_LEN*BATCH)     // 8192
#define M_DEC (TDEC*BATCH)      // 8128
#define HSZ   (BATCH*HID)       // elems in one h buffer

typedef __attribute__((ext_vector_type(16))) __bf16 v16bf;
typedef __attribute__((ext_vector_type(8)))  float  v8f;

union Frag { v16bf v; uint4 q[2]; };

__device__ inline float bf2f(unsigned short u){
  unsigned int x = ((unsigned int)u) << 16; float f; __builtin_memcpy(&f,&x,4); return f;
}
__device__ inline unsigned short f2bf(float f){
  unsigned int x; __builtin_memcpy(&x,&f,4);
  x += 0x7FFFu + ((x>>16)&1u);                 // round-to-nearest-even
  return (unsigned short)(x>>16);
}
__device__ inline float sigm(float x){ return 1.0f/(1.0f+__expf(-x)); }

__device__ inline v8f wmma_bf16(v16bf a, v16bf b, v8f c){
  // D = A(16x32 bf16) x B(32x16 bf16) + C(16x16 f32)
  return __builtin_amdgcn_wmma_f32_16x16x32_bf16(false, a, false, b, (short)0, c, false, false);
}

// A fragment from row-major A[M,K] bf16.  ISA layout: lane l holds row (l&15);
// half=l>>4 selects K sub-pattern: elems 0..7 -> k = kb + half*8 + e,
// elems 8..15 -> k = kb + 16 + half*8 + (e-8).  => two contiguous 16B loads.
__device__ inline v16bf load_a_frag(const unsigned short* Arow, int kb, int half){
  Frag f;
  f.q[0] = *(const uint4*)(Arow + kb + half*8);
  f.q[1] = *(const uint4*)(Arow + kb + 16 + half*8);
  return f.v;
}
// B fragment from row-major W[N,K] bf16 (B = W^T): lane l holds col n=(l&15),
// same K pattern as A: elems 0..7 -> k = kb + half*8 + e, elems 8..15 -> k = kb + 16 + half*8 + (e-8).
__device__ inline v16bf load_b_frag(const unsigned short* Wrow, int kb, int half){
  Frag f;
  f.q[0] = *(const uint4*)(Wrow + kb + half*8);
  f.q[1] = *(const uint4*)(Wrow + kb + 16 + half*8); return f.v;
}

// ---------------- small prep kernels ----------------
__global__ void f32_to_bf16_kernel(const float* __restrict__ in,
                                   unsigned short* __restrict__ out, int n){
  int i = blockIdx.x*blockDim.x + threadIdx.x;
  if (i < n) out[i] = f2bf(in[i]);
}

__global__ void bias_sum_kernel(const float* __restrict__ a, const float* __restrict__ b,
                                float* __restrict__ out, int n){
  int i = blockIdx.x*blockDim.x + threadIdx.x;
  if (i < n) out[i] = a[i] + b[i];
}

// gather embedding rows + cast to bf16; X row order == (t*B + b) flat token order
__global__ void gather_cast_kernel(const int* __restrict__ tok,
                                   const float* __restrict__ emb,
                                   unsigned short* __restrict__ X, int nrows){
  int row = blockIdx.x; if (row >= nrows) return;
  int t = tok[row];
  const float* e = emb + (size_t)t * KDIM;
  unsigned short* x = X + (size_t)row * KDIM;
  for (int i = threadIdx.x; i < KDIM; i += blockDim.x) x[i] = f2bf(e[i]);
}

__global__ void init_state_kernel(unsigned short* h0buf, unsigned short* h1buf, int* bar){
  int i = blockIdx.x*blockDim.x + threadIdx.x;
  if (i < 2*HSZ){ h0buf[i] = 0; h1buf[i] = 0; }
  if (i < 2) bar[i] = 0;
}

// ---------------- big WMMA GEMM ----------------
// C[M, Nreal](f32 or bf16) = A[M,K](bf16) @ W[Nreal,K]^T (bf16) + bias[n]
// block = 256 thr = 8 waves (2 M x 4 N); wave tile = 2x4 of 16x16; block tile 64x256.
// M must be a multiple of 64 (true for 8192 / 8128); N is bounds-checked.
__global__ __launch_bounds__(256)
void gemm_bf16_kernel(const unsigned short* __restrict__ A,
                      const unsigned short* __restrict__ W,
                      const float* __restrict__ bias,
                      void* __restrict__ out,
                      int Nreal, int K, int out_bf16){
  const int lane = threadIdx.x & 31;
  const int wave = threadIdx.x >> 5;
  const int wm = wave >> 2, wn = wave & 3;
  const int half = lane >> 4, lm = lane & 15;
  const int Mbase = blockIdx.y * 64;
  const int Nbase = blockIdx.x * 256;

  const unsigned short* Arow[2];
  #pragma unroll
  for (int mt = 0; mt < 2; ++mt)
    Arow[mt] = A + (size_t)(Mbase + wm*32 + mt*16 + lm) * K;

  int ncol[4]; const unsigned short* Wrow[4];
  #pragma unroll
  for (int nt = 0; nt < 4; ++nt){
    int n = Nbase + wn*64 + nt*16 + lm;
    ncol[nt] = n;
    int nc = n < Nreal ? n : (Nreal - 1);
    Wrow[nt] = W + (size_t)nc * K;
  }

  v8f acc[2][4];
  #pragma unroll
  for (int mt = 0; mt < 2; ++mt)
    #pragma unroll
    for (int nt = 0; nt < 4; ++nt){
      float bv = (bias && ncol[nt] < Nreal) ? bias[ncol[nt]] : 0.0f;
      #pragma unroll
      for (int j = 0; j < 8; ++j) acc[mt][nt][j] = bv;
    }

  const int kts = K >> 5;
  for (int kt = 0; kt < kts; ++kt){
    int kb = kt * 32;
    if (kt + 1 < kts){
      __builtin_prefetch(Arow[0] + kb + 32, 0, 1);     // -> global_prefetch_b8
      __builtin_prefetch(Wrow[0] + kb + 32, 0, 1);
    }
    v16bf a[2];
    #pragma unroll
    for (int mt = 0; mt < 2; ++mt) a[mt] = load_a_frag(Arow[mt], kb, half);
    #pragma unroll
    for (int nt = 0; nt < 4; ++nt){
      v16bf b = load_b_frag(Wrow[nt], kb, half);
      #pragma unroll
      for (int mt = 0; mt < 2; ++mt)
        acc[mt][nt] = wmma_bf16(a[mt], b, acc[mt][nt]);
    }
  }

  // C/D layout: VGPR j, lane l -> row = j + (l>>4)*8, col = l&15 (within tile)
  #pragma unroll
  for (int mt = 0; mt < 2; ++mt)
    #pragma unroll
    for (int nt = 0; nt < 4; ++nt){
      int n = ncol[nt];
      if (n >= Nreal) continue;
      #pragma unroll
      for (int j = 0; j < 8; ++j){
        int r = Mbase + wm*32 + mt*16 + j + half*8;
        size_t idx = (size_t)r * Nreal + n;
        if (out_bf16) ((unsigned short*)out)[idx] = f2bf(acc[mt][nt][j]);
        else          ((float*)out)[idx]          = acc[mt][nt][j];
      }
    }
}

// ---------------- persistent LSTM recurrence ----------------
__device__ inline void grid_barrier(int* bar, int nblk, int* sgen){
  __syncthreads();
  if (threadIdx.x == 0){
    __threadfence();
    int g = *sgen;
    if (atomicAdd(&bar[0], 1) == nblk - 1){
      bar[0] = 0;
      __threadfence();
      atomicExch(&bar[1], g + 1);
    } else {
      while (atomicAdd(&bar[1], 0) < g + 1) __builtin_amdgcn_s_sleep(2);
    }
    *sgen = g + 1;
  }
  __syncthreads();
}

// One LSTM layer step for this wave's (mt,jt) patch.  The wave owns gates
// i,f,g,o for batch rows [mt*16,mt*16+16) x hidden cols [jt*16,jt*16+16),
// so cell state c stays in registers across the whole sequence.
__device__ void lstm_layer_step(v8f* cptr,
                                const unsigned short* preX,   // [64,2048] bf16 or null
                                const float* bias,            // [2048] or null
                                const unsigned short* hinA, const unsigned short* WA, // h @ WA^T
                                const unsigned short* hinB, const unsigned short* WB, // optional 2nd GEMM
                                unsigned short* hout, unsigned short* ysout,
                                int mt, int jt, int lane){
  const int half = lane >> 4, lm = lane & 15;
  v8f acc[4];
  if (preX){
    #pragma unroll
    for (int g = 0; g < 4; ++g){
      int n = g*HID + jt*16 + lm;
      #pragma unroll
      for (int j = 0; j < 8; ++j){
        int m = mt*16 + j + half*8;
        acc[g][j] = bf2f(preX[(size_t)m*G4H + n]);
      }
    }
  } else {
    #pragma unroll
    for (int g = 0; g < 4; ++g){
      float bv = bias[g*HID + jt*16 + lm];
      #pragma unroll
      for (int j = 0; j < 8; ++j) acc[g][j] = bv;
    }
  }

  const unsigned short* ArowA = hinA + (size_t)(mt*16 + lm) * HID;
  for (int kt = 0; kt < (HID >> 5); ++kt){
    int kb = kt * 32;
    v16bf a = load_a_frag(ArowA, kb, half);
    #pragma unroll
    for (int g = 0; g < 4; ++g){
      const unsigned short* Wr = WA + (size_t)(g*HID + jt*16 + lm) * HID;
      acc[g] = wmma_bf16(a, load_b_frag(Wr, kb, half), acc[g]);
    }
  }
  if (WB){
    const unsigned short* ArowB = hinB + (size_t)(mt*16 + lm) * HID;
    for (int kt = 0; kt < (HID >> 5); ++kt){
      int kb = kt * 32;
      v16bf a = load_a_frag(ArowB, kb, half);
      #pragma unroll
      for (int g = 0; g < 4; ++g){
        const unsigned short* Wr = WB + (size_t)(g*HID + jt*16 + lm) * HID;
        acc[g] = wmma_bf16(a, load_b_frag(Wr, kb, half), acc[g]);
      }
    }
  }

  v8f c = *cptr, h;
  #pragma unroll
  for (int j = 0; j < 8; ++j){
    float iv = sigm(acc[0][j]);
    float fv = sigm(acc[1][j]);
    float gv = tanhf(acc[2][j]);
    float ov = sigm(acc[3][j]);
    float cv = fv * c[j] + iv * gv;
    c[j] = cv;
    h[j] = ov * tanhf(cv);
  }
  *cptr = c;

  int n = jt*16 + lm;
  #pragma unroll
  for (int j = 0; j < 8; ++j){
    unsigned short hb = f2bf(h[j]);
    int m = mt*16 + j + half*8;
    hout[(size_t)m*HID + n] = hb;
    if (ysout) ysout[(size_t)m*HID + n] = hb;
  }
}

__global__ __launch_bounds__(128)
void lstm_seq_kernel(const unsigned short* __restrict__ preXe,  // [8192,2048] bf16 (incl. bias)
                     const unsigned short* __restrict__ preXd,  // [8128,2048] bf16 (incl. bias)
                     const unsigned short* Whh_e0, const unsigned short* Wih_e1, const unsigned short* Whh_e1,
                     const unsigned short* Whh_d0, const unsigned short* Wih_d1, const unsigned short* Whh_d1,
                     const float* b_e1, const float* b_d1,
                     unsigned short* h0buf, unsigned short* h1buf,   // [2][64*512] bf16, double-buffered
                     unsigned short* ys,                              // [8128,512] bf16
                     int* bar){
  __shared__ int sgen;
  if (threadIdx.x == 0) sgen = 0;
  __syncthreads();

  const int lane = threadIdx.x & 31;
  const int wid  = blockIdx.x * (blockDim.x >> 5) + (threadIdx.x >> 5); // 0..127
  const int mt = wid & 3, jt = wid >> 2;
  const int nblk = gridDim.x;

  v8f c0, c1;
  #pragma unroll
  for (int j = 0; j < 8; ++j){ c0[j] = 0.0f; c1[j] = 0.0f; }

  for (int t = 0; t < S_LEN + TDEC; ++t){
    const bool enc = t < S_LEN;
    const int cur = t & 1, prev = cur ^ 1;
    const unsigned short* preX = enc ? preXe + (size_t)t * BATCH * G4H
                                     : preXd + (size_t)(t - S_LEN) * BATCH * G4H;
    // layer 0:  g = preX[t] + h0_prev @ Whh0^T  (cell state c0 stays in regs)
    lstm_layer_step(&c0, preX, nullptr,
                    h0buf + (size_t)prev*HSZ, enc ? Whh_e0 : Whh_d0,
                    nullptr, nullptr,
                    h0buf + (size_t)cur*HSZ, nullptr, mt, jt, lane);
    grid_barrier(bar, nblk, &sgen);
    // layer 1:  g = b1 + h0_cur @ Wih1^T + h1_prev @ Whh1^T
    unsigned short* ysout = enc ? nullptr : ys + (size_t)(t - S_LEN) * BATCH * HID;
    lstm_layer_step(&c1, nullptr, enc ? b_e1 : b_d1,
                    h0buf + (size_t)cur*HSZ,  enc ? Wih_e1 : Wih_d1,
                    h1buf + (size_t)prev*HSZ, enc ? Whh_e1 : Whh_d1,
                    h1buf + (size_t)cur*HSZ, ysout, mt, jt, lane);
    grid_barrier(bar, nblk, &sgen);
  }
}

// ---------------- host launcher ----------------
extern "C" void kernel_launch(void* const* d_in, const int* in_sizes, int n_in,
                              void* d_out, int out_size, void* d_ws, size_t ws_size,
                              hipStream_t stream){
  const int*   src     = (const int*)  d_in[0];
  const int*   trg     = (const int*)  d_in[1];
  const float* enc_emb = (const float*)d_in[3];
  const float* dec_emb = (const float*)d_in[4];
  const float* enc_Wih = (const float*)d_in[5];
  const float* enc_Whh = (const float*)d_in[6];
  const float* enc_bih = (const float*)d_in[7];
  const float* enc_bhh = (const float*)d_in[8];
  const float* dec_Wih = (const float*)d_in[9];
  const float* dec_Whh = (const float*)d_in[10];
  const float* dec_bih = (const float*)d_in[11];
  const float* dec_bhh = (const float*)d_in[12];
  const float* out_W   = (const float*)d_in[13];
  const float* out_b   = (const float*)d_in[14];
  float* logits = (float*)d_out;

  // workspace carve-up
  size_t off = 0;
  auto carve = [&](size_t bytes)->char*{
    char* p = (char*)d_ws + off;
    off = (off + bytes + 255) & ~(size_t)255;
    return p;
  };
  unsigned short* Xe    = (unsigned short*)carve((size_t)M_ENC*KDIM*2);
  unsigned short* Xd    = (unsigned short*)carve((size_t)M_DEC*KDIM*2);
  unsigned short* We_ih = (unsigned short*)carve((size_t)2*G4H*KDIM*2);
  unsigned short* We_hh = (unsigned short*)carve((size_t)2*G4H*KDIM*2);
  unsigned short* Wd_ih = (unsigned short*)carve((size_t)2*G4H*KDIM*2);
  unsigned short* Wd_hh = (unsigned short*)carve((size_t)2*G4H*KDIM*2);
  unsigned short* Wo    = (unsigned short*)carve((size_t)VOCAB*KDIM*2);
  float* bsum_e         = (float*)carve((size_t)2*G4H*4);
  float* bsum_d         = (float*)carve((size_t)2*G4H*4);
  unsigned short* preXe = (unsigned short*)carve((size_t)M_ENC*G4H*2);
  unsigned short* preXd = (unsigned short*)carve((size_t)M_DEC*G4H*2);
  unsigned short* h0buf = (unsigned short*)carve((size_t)2*HSZ*2);
  unsigned short* h1buf = (unsigned short*)carve((size_t)2*HSZ*2);
  unsigned short* ys    = (unsigned short*)carve((size_t)M_DEC*HID*2);
  int* bar              = (int*)carve(256);
  (void)ws_size; (void)in_sizes; (void)n_in; (void)out_size;

  // 1) weight/bias prep (f32 -> bf16 row-major; no swizzle needed)
  int nw = 2*G4H*KDIM;
  f32_to_bf16_kernel<<<(nw+255)/256, 256, 0, stream>>>(enc_Wih, We_ih, nw);
  f32_to_bf16_kernel<<<(nw+255)/256, 256, 0, stream>>>(enc_Whh, We_hh, nw);
  f32_to_bf16_kernel<<<(nw+255)/256, 256, 0, stream>>>(dec_Wih, Wd_ih, nw);
  f32_to_bf16_kernel<<<(nw+255)/256, 256, 0, stream>>>(dec_Whh, Wd_hh, nw);
  int no = VOCAB*KDIM;
  f32_to_bf16_kernel<<<(no+255)/256, 256, 0, stream>>>(out_W, Wo, no);
  bias_sum_kernel<<<(2*G4H+255)/256, 256, 0, stream>>>(enc_bih, enc_bhh, bsum_e, 2*G4H);
  bias_sum_kernel<<<(2*G4H+255)/256, 256, 0, stream>>>(dec_bih, dec_bhh, bsum_d, 2*G4H);

  // 2) embedding gathers (+bf16 cast)
  gather_cast_kernel<<<M_ENC, 256, 0, stream>>>(src, enc_emb, Xe, M_ENC);
  gather_cast_kernel<<<M_DEC, 256, 0, stream>>>(trg, dec_emb, Xd, M_DEC); // trg[:-1] == first 127*64 tokens

  // 3) zero h state + barrier (every launch -> deterministic / graph-safe)
  init_state_kernel<<<(2*HSZ+255)/256, 256, 0, stream>>>(h0buf, h1buf, bar);

  // 4) time-parallel layer-0 input GEMMs (bias folded in), bf16 out
  gemm_bf16_kernel<<<dim3(G4H/256, M_ENC/64), 256, 0, stream>>>(Xe, We_ih, bsum_e, preXe, G4H, KDIM, 1);
  gemm_bf16_kernel<<<dim3(G4H/256, M_DEC/64), 256, 0, stream>>>(Xd, Wd_ih, bsum_d, preXd, G4H, KDIM, 1);

  // 5) persistent recurrence: encoder 128 steps then decoder 127 steps,
  //    cell state in registers, h double-buffered through L2
  lstm_seq_kernel<<<32, 128, 0, stream>>>(preXe, preXd,
      We_hh,                 We_ih + (size_t)G4H*KDIM, We_hh + (size_t)G4H*KDIM,
      Wd_hh,                 Wd_ih + (size_t)G4H*KDIM, Wd_hh + (size_t)G4H*KDIM,
      bsum_e + G4H, bsum_d + G4H,
      h0buf, h1buf, ys, bar);

  // 6) output projection: [8128,512] @ [512,16000] + out_b -> f32 logits
  gemm_bf16_kernel<<<dim3((VOCAB+255)/256, M_DEC/64), 256, 0, stream>>>(ys, Wo, out_b, logits, VOCAB, KDIM, 0);
}
